// SmallWorldScaleFreeGraphFusion_83975200571946
// MI455X (gfx1250) — matrix operand order repaired and checked by reading, not compiled, not run on hardware
//
#include <hip/hip_runtime.h>

// ---------------------------------------------------------------------------
// CDNA5 (gfx1250) WMMA types / helpers
// ---------------------------------------------------------------------------
typedef __attribute__((ext_vector_type(16))) __bf16 v16bf;
typedef __attribute__((ext_vector_type(8)))  __bf16 v8bf;
typedef __attribute__((ext_vector_type(4)))  __bf16 v4bf;
typedef __attribute__((ext_vector_type(8)))  float  v8f;

#define BATCH   64
#define NNODE   320
#define HID     512
#define HEADS   8
#define HDIM    64

__device__ __forceinline__ v8f wmma_bf16(v16bf a, v16bf b, v8f c) {
  return __builtin_amdgcn_wmma_f32_16x16x32_bf16(
      /*neg_a=*/false, a, /*neg_b=*/false, b,
      /*c_mod=*/(short)0, c, /*reuse_a=*/false, /*reuse_b=*/false);
}

// A fragment (16x32) from row-major bf16 [M,lda]. ISA 7.12.2 16-bit A layout:
// lane {hs,r}: elems 0..7 = K kbase+hs*8.., elems 8..15 = K kbase+16+hs*8..
// => two 16-byte contiguous loads per lane.
__device__ __forceinline__ v16bf load_a_bf(const __bf16* __restrict__ Arow0,
                                           int lda, int kbase, int lane) {
  const int r  = lane & 15;
  const int hs = (lane >> 4) & 1;
  const __bf16* p = Arow0 + (size_t)r * lda + kbase + hs * 8;
  v8bf lo = *(const v8bf*)p;
  v8bf hi = *(const v8bf*)(p + 16);
  return __builtin_shufflevector(lo, hi, 0, 1, 2, 3, 4, 5, 6, 7,
                                 8, 9, 10, 11, 12, 13, 14, 15);
}

// B fragment (32x16) from TRANSPOSED bf16 weights WT[N][K]:
// lane {hs,n}: elems 0..7 = WT[col+n][kbase+hs*8..], elems 8..15 = WT[col+n][kbase+16+hs*8..] -> two 16-byte loads.
__device__ __forceinline__ v16bf load_b_T(const __bf16* __restrict__ WT,
                                          int ldk, int kbase, int col, int lane) {
  const int n  = lane & 15;
  const int hs = (lane >> 4) & 1;
  const __bf16* p = WT + (size_t)(col + n) * ldk + kbase + hs * 8;
  v8bf lo = *(const v8bf*)p;
  v8bf hi = *(const v8bf*)(p + 16);
  return __builtin_shufflevector(lo, hi, 0, 1, 2, 3, 4, 5, 6, 7,
                                 8, 9, 10, 11, 12, 13, 14, 15);
}

// ---------------------------------------------------------------------------
// One-time conversion kernels
// ---------------------------------------------------------------------------
// f32 -> bf16, 4 elements/thread (b128 in, b64 out)
__global__ void __launch_bounds__(256) cvt_bf16_x4(const float* __restrict__ src,
                                                   __bf16* __restrict__ dst, int n4) {
  const int i = blockIdx.x * blockDim.x + threadIdx.x;
  if (i >= n4) return;
  const float4 v = ((const float4*)src)[i];
  v4bf o;
  o[0] = (__bf16)v.x; o[1] = (__bf16)v.y; o[2] = (__bf16)v.z; o[3] = (__bf16)v.w;
  ((v4bf*)dst)[i] = o;
}

// pad columns: src[R,sC] f32 -> dst[R,dC] bf16 (zero tail)
__global__ void __launch_bounds__(256) pad_cvt(const float* __restrict__ src,
                                               __bf16* __restrict__ dst,
                                               int R, int sC, int dC) {
  const int idx = blockIdx.x * blockDim.x + threadIdx.x;
  if (idx >= R * dC) return;
  const int r = idx / dC;
  const int c = idx - r * dC;
  dst[idx] = (__bf16)((c < sC) ? src[(size_t)r * sC + c] : 0.0f);
}

// transpose+pad+convert: src[srcK, N] f32 -> dst[N, Kd] bf16 (zero pad k>=srcK)
__global__ void __launch_bounds__(256) transpose_cvt(const float* __restrict__ src,
                                                     __bf16* __restrict__ dst,
                                                     int srcK, int N, int Kd) {
  const int idx = blockIdx.x * blockDim.x + threadIdx.x;
  if (idx >= N * Kd) return;
  const int n = idx / Kd;
  const int k = idx - n * Kd;
  dst[idx] = (__bf16)((k < srcK) ? src[(size_t)k * N + n] : 0.0f);
}

// Wg [2,8,512,64] f32 -> WgT [2,8,64,512] bf16
__global__ void __launch_bounds__(256) transpose_wg(const float* __restrict__ Wg,
                                                    __bf16* __restrict__ WgT) {
  const int idx = blockIdx.x * blockDim.x + threadIdx.x;  // 2*8*64*512 exactly
  const int d   = idx & (HID - 1);
  const int f   = (idx >> 9) & (HDIM - 1);
  const int lh  = idx >> 15;                               // l*8+head
  WgT[idx] = (__bf16)Wg[(((size_t)lh * HID) + d) * HDIM + f];
}

// ---------------------------------------------------------------------------
// Kernel 1: modality projection  Xbf[b, nodeOffset+n, :] = A @ W   (bf16 in/out)
// A: bf16 [M,K]; WT: bf16 [512,K]; wave -> 32(M) x 64(N) tile; K % 32 == 0.
// ---------------------------------------------------------------------------
template <int K>
__global__ void __launch_bounds__(256) proj_gemm(const __bf16* __restrict__ A,
                                                 const __bf16* __restrict__ WT,
                                                 __bf16* __restrict__ Xbf,
                                                 int rowsPerB, int nodeOffset) {
  const int gwave    = (blockIdx.x * blockDim.x + threadIdx.x) >> 5;
  const int lane     = threadIdx.x & 31;
  const int mTile    = gwave >> 3;
  const int n0       = (gwave & 7) * 64;
  const int row_base = mTile * 32;
  const __bf16* Arow0 = A + (size_t)row_base * K;
  const __bf16* Arow1 = Arow0 + (size_t)16 * K;

  v8f acc[2][4] = {};
  for (int kb = 0; kb < K; kb += 32) {
    __builtin_prefetch(Arow0 + (size_t)(lane & 15) * K + kb + 32, 0, 3);
    v16bf a0 = load_a_bf(Arow0, K, kb, lane);
    v16bf a1 = load_a_bf(Arow1, K, kb, lane);
#pragma unroll
    for (int t = 0; t < 4; ++t) {
      v16bf b = load_b_T(WT, K, kb, n0 + t * 16, lane);
      acc[0][t] = wmma_bf16(a0, b, acc[0][t]);
      acc[1][t] = wmma_bf16(a1, b, acc[1][t]);
    }
  }

  const int ncol = lane & 15;
  const int hs   = (lane >> 4) & 1;
#pragma unroll
  for (int mt = 0; mt < 2; ++mt) {
#pragma unroll
    for (int t = 0; t < 4; ++t) {
#pragma unroll
      for (int r = 0; r < 8; ++r) {
        const int m  = row_base + mt * 16 + hs * 8 + r;   // D: VGPR r -> M=hs*8+r
        const int bb = m / rowsPerB;
        const int nl = m - bb * rowsPerB;
        Xbf[(((size_t)bb * NNODE) + nodeOffset + nl) * HID + n0 + t * 16 + ncol] =
            (__bf16)acc[mt][t][r];
      }
    }
  }
}

// ---------------------------------------------------------------------------
// Kernel 2: head projection, TRANSPOSED output.
// HT[b][head][f][node] = (Xbf[b,node,:] @ Wg[l,head,:,:])^T
// Xbf: [B*320, 512]; WgTl: [8][64][512]; HT: [B*8][64][320]
// D-layout gives 8 consecutive nodes per lane -> packed v8bf stores.
// ---------------------------------------------------------------------------
__global__ void __launch_bounds__(256) hidden_gemm(const __bf16* __restrict__ Xbf,
                                                   const __bf16* __restrict__ WgTl,
                                                   __bf16* __restrict__ HT) {
  const int gwave    = (blockIdx.x * blockDim.x + threadIdx.x) >> 5;
  const int lane     = threadIdx.x & 31;
  const int mTile    = gwave >> 3;
  const int n0       = (gwave & 7) * 64;       // 64-wide tile == exactly one head
  const int head     = n0 >> 6;
  const int row_base = mTile * 32;
  const __bf16* Arow0 = Xbf + (size_t)row_base * HID;
  const __bf16* Arow1 = Arow0 + (size_t)16 * HID;
  const __bf16* WhT   = WgTl + (size_t)head * HDIM * HID;   // [64][512]

  v8f acc[2][4] = {};
  for (int kb = 0; kb < HID; kb += 32) {
    __builtin_prefetch(Arow0 + (size_t)(lane & 15) * HID + kb + 32, 0, 3);
    v16bf a0 = load_a_bf(Arow0, HID, kb, lane);
    v16bf a1 = load_a_bf(Arow1, HID, kb, lane);
#pragma unroll
    for (int t = 0; t < 4; ++t) {
      v16bf b = load_b_T(WhT, HID, kb, t * 16, lane);
      acc[0][t] = wmma_bf16(a0, b, acc[0][t]);
      acc[1][t] = wmma_bf16(a1, b, acc[1][t]);
    }
  }

  const int ncol = lane & 15;
  const int hs   = (lane >> 4) & 1;
#pragma unroll
  for (int mt = 0; mt < 2; ++mt) {
    const int m0 = row_base + mt * 16 + hs * 8;   // 8-node run, never crosses batch
    const int bb = m0 / NNODE;
    const int i0 = m0 - bb * NNODE;
#pragma unroll
    for (int t = 0; t < 4; ++t) {
      v8bf o;
#pragma unroll
      for (int r = 0; r < 8; ++r) o[r] = (__bf16)acc[mt][t][r];
      *(v8bf*)(HT + ((((size_t)bb * HEADS + head) * HDIM) + t * 16 + ncol) * NNODE + i0) = o;
    }
  }
}

// ---------------------------------------------------------------------------
// Kernel 3: fi = h.a_src, fj = h.a_dst from transposed HT (coalesced over nodes)
// ---------------------------------------------------------------------------
__global__ void __launch_bounds__(320) edge_coef(const __bf16* __restrict__ HT,
                                                 const float* __restrict__ Asrc,
                                                 const float* __restrict__ Adst,
                                                 float* __restrict__ Fs,
                                                 float* __restrict__ Fd) {
  const int bh   = blockIdx.x;            // b*HEADS + head
  const int n    = threadIdx.x;           // 0..319
  const int head = bh & 7;
  const __bf16* Hb = HT + (size_t)bh * HDIM * NNODE;
  const float* as = Asrc + head * HDIM;
  const float* ad = Adst + head * HDIM;
  float s = 0.0f, d = 0.0f;
#pragma unroll
  for (int f = 0; f < HDIM; ++f) {
    const float v = (float)Hb[(size_t)f * NNODE + n];
    s += v * as[f];
    d += v * ad[f];
  }
  Fs[(size_t)bh * NNODE + n] = s;
  Fd[(size_t)bh * NNODE + n] = d;
}

// ---------------------------------------------------------------------------
// Kernel 4: fused masked-softmax attention + message passing (att @ h).
// One block (128 thr, 4 waves) per (b*head, 16-row tile). Probabilities live
// only in LDS (f32 for softmax math, bf16 copy for WMMA A-fragments).
// ---------------------------------------------------------------------------
__global__ void __launch_bounds__(128) attn_kernel(const __bf16* __restrict__ HT,
                                                   const float* __restrict__ Fs,
                                                   const float* __restrict__ Fd,
                                                   const int*   __restrict__ adj,
                                                   const float* __restrict__ elog,
                                                   __bf16* __restrict__ outbf,
                                                   float*  __restrict__ outf,
                                                   int is_final) {
  __shared__ float  att[16][NNODE];       // 20 KB
  __shared__ __bf16 attb[16][NNODE];      // 10 KB
  __shared__ float  red[16][8];

  const int bh    = blockIdx.x / 20;      // b*HEADS + head
  const int iTile = blockIdx.x % 20;
  const int b     = bh >> 3;
  const int head  = bh & 7;
  const int tid   = threadIdx.x;

  const float* FsRow = Fs + (size_t)bh * NNODE;
  const float* FdRow = Fd + (size_t)bh * NNODE;

  // ---- phase 1: masked leaky-relu scores + softmax (8 threads per row) ----
  const int row = tid >> 3;
  const int sub = tid & 7;
  const int i   = iTile * 16 + row;
  const float fi = FsRow[i];
  const int*   adjRow  = adj  + (size_t)i * NNODE;
  const float* elogRow = elog + (size_t)i * NNODE;
  const float NEG_INF = -__builtin_inff();

  float mx = NEG_INF;
#pragma unroll 4
  for (int j = sub; j < NNODE; j += 8) {
    float e;
    if (adjRow[j] != 0) {
      float t = fi + FdRow[j];
      t = (t > 0.0f) ? t : 0.2f * t;      // leaky_relu(., 0.2)
      e = t + elogRow[j];
    } else {
      e = NEG_INF;
    }
    att[row][j] = e;
    mx = fmaxf(mx, e);
  }
  red[row][sub] = mx;
  __syncthreads();
  float rmax = NEG_INF;
#pragma unroll
  for (int k = 0; k < 8; ++k) rmax = fmaxf(rmax, red[row][k]);
  __syncthreads();

  float sum = 0.0f;
#pragma unroll 4
  for (int j = sub; j < NNODE; j += 8) {
    const float e = att[row][j];
    const float p = (e > NEG_INF && rmax > NEG_INF) ? __expf(e - rmax) : 0.0f;
    att[row][j] = p;
    sum += p;
  }
  red[row][sub] = sum;
  __syncthreads();
  float rsum = 0.0f;
#pragma unroll
  for (int k = 0; k < 8; ++k) rsum += red[row][k];
  const float inv = (rsum > 0.0f) ? (1.0f / rsum) : 0.0f;   // nan_to_num rows -> 0
  __syncthreads();
#pragma unroll 4
  for (int j = sub; j < NNODE; j += 8) attb[row][j] = (__bf16)(att[row][j] * inv);
  __syncthreads();

  // ---- phase 2: out[16,64] = att[16,320] @ H[320,64] = att @ HT^T ----
  const int wv   = tid >> 5;
  const int lane = tid & 31;
  const int f0   = wv * 16;
  const __bf16* HTbh = HT + (size_t)bh * HDIM * NNODE;
  const int rr = lane & 15;
  const int hs = (lane >> 4) & 1;

  v8f acc = {};
#pragma unroll
  for (int kb = 0; kb < NNODE; kb += 32) {
    const __bf16* p = &attb[rr][kb + hs * 8];
    v8bf lo = *(const v8bf*)p;            // ds_load_b128
    v8bf hi = *(const v8bf*)(p + 16);
    v16bf a = __builtin_shufflevector(lo, hi, 0, 1, 2, 3, 4, 5, 6, 7,
                                      8, 9, 10, 11, 12, 13, 14, 15);
    v16bf bf = load_b_T(HTbh, NNODE, kb, f0, lane);
    acc = wmma_bf16(a, bf, acc);
  }

  const int ncol = lane & 15;
#pragma unroll
  for (int r = 0; r < 8; ++r) {
    const int irow = iTile * 16 + hs * 8 + r;
    float v = acc[r];
    const size_t o = (((size_t)b * NNODE) + irow) * HID + head * HDIM + f0 + ncol;
    if (is_final) {
      outf[o] = v;                                        // layer 1 -> f32 d_out
    } else {
      v = (v > 0.0f) ? v : (__expf(v) - 1.0f);            // ELU between layers
      outbf[o] = (__bf16)v;                               // layer 0 -> bf16 X
    }
  }
}

// ---------------------------------------------------------------------------
// Orchestration
// ---------------------------------------------------------------------------
extern "C" void kernel_launch(void* const* d_in, const int* in_sizes, int n_in,
                              void* d_out, int out_size, void* d_ws, size_t ws_size,
                              hipStream_t stream) {
  (void)in_sizes; (void)n_in; (void)out_size; (void)ws_size;

  const float* text     = (const float*)d_in[0];   // [64,256,768]
  const float* visual   = (const float*)d_in[1];   // [64,32,47]
  const float* acoustic = (const float*)d_in[2];   // [64,32,74]
  const int*   adj      = (const int*)  d_in[3];   // [320,320] int32
  const float* Wt       = (const float*)d_in[4];   // [768,512]
  const float* Wv       = (const float*)d_in[5];   // [47,512]
  const float* Wa       = (const float*)d_in[6];   // [74,512]
  const float* Wg       = (const float*)d_in[7];   // [2,8,512,64]
  const float* a_src    = (const float*)d_in[8];   // [2,8,64]
  const float* a_dst    = (const float*)d_in[9];   // [2,8,64]
  const float* elog     = (const float*)d_in[10];  // [320,320]
  float* out = (float*)d_out;                      // [64,320,512] f32

  // ---- workspace carve-up (256-byte aligned regions) ----
  char* base = (char*)d_ws;
  auto carve = [&](size_t bytes) {
    char* p = base;
    base += (bytes + 255) & ~(size_t)255;
    return p;
  };
  __bf16* Xbf   = (__bf16*)carve((size_t)BATCH * NNODE * HID * 2);          // 21 MB
  __bf16* HT    = (__bf16*)carve((size_t)BATCH * HEADS * HDIM * NNODE * 2); // 21 MB
  float*  Fs    = (float*) carve((size_t)BATCH * HEADS * NNODE * 4);
  float*  Fd    = (float*) carve((size_t)BATCH * HEADS * NNODE * 4);
  __bf16* Tbf   = (__bf16*)carve((size_t)BATCH * 256 * 768 * 2);            // 25 MB
  __bf16* Vbf   = (__bf16*)carve((size_t)BATCH * 32 * 64 * 2);
  __bf16* Abf   = (__bf16*)carve((size_t)BATCH * 32 * 96 * 2);
  __bf16* WtT   = (__bf16*)carve((size_t)HID * 768 * 2);
  __bf16* WvT   = (__bf16*)carve((size_t)HID * 64 * 2);
  __bf16* WaT   = (__bf16*)carve((size_t)HID * 96 * 2);
  __bf16* WgT   = (__bf16*)carve((size_t)2 * HEADS * HDIM * HID * 2);

  // ---- one-time conversions / transposes (all bf16 from here on) ----
  cvt_bf16_x4<<<(BATCH * 256 * 768 / 4 + 255) / 256, 256, 0, stream>>>(
      text, Tbf, BATCH * 256 * 768 / 4);
  pad_cvt<<<(BATCH * 32 * 64 + 255) / 256, 256, 0, stream>>>(visual, Vbf,
                                                             BATCH * 32, 47, 64);
  pad_cvt<<<(BATCH * 32 * 96 + 255) / 256, 256, 0, stream>>>(acoustic, Abf,
                                                             BATCH * 32, 74, 96);
  transpose_cvt<<<(HID * 768 + 255) / 256, 256, 0, stream>>>(Wt, WtT, 768, HID, 768);
  transpose_cvt<<<(HID * 64 + 255) / 256, 256, 0, stream>>>(Wv, WvT, 47, HID, 64);
  transpose_cvt<<<(HID * 96 + 255) / 256, 256, 0, stream>>>(Wa, WaT, 74, HID, 96);
  transpose_wg<<<(2 * HEADS * HDIM * HID + 255) / 256, 256, 0, stream>>>(Wg, WgT);

  // ---- modality projections into Xbf (nodes 0..255 | 256..287 | 288..319) ----
  proj_gemm<768><<<512, 256, 0, stream>>>(Tbf, WtT, Xbf, 256, 0);   // 4096 waves
  proj_gemm<64><<<64, 256, 0, stream>>>(Vbf, WvT, Xbf, 32, 256);    // 512 waves
  proj_gemm<96><<<64, 256, 0, stream>>>(Abf, WaT, Xbf, 32, 288);

  for (int l = 0; l < 2; ++l) {
    const __bf16* WgTl = WgT + (size_t)l * HEADS * HDIM * HID;
    // M = 20480 -> 640 m-tiles(32) * 8 n-tiles = 5120 waves = 640 blocks
    hidden_gemm<<<640, 256, 0, stream>>>(Xbf, WgTl, HT);
    // one block of 320 threads per (b,head)
    edge_coef<<<BATCH * HEADS, 320, 0, stream>>>(HT, a_src + l * HEADS * HDIM,
                                                 a_dst + l * HEADS * HDIM, Fs, Fd);
    // (B*HEADS) * (320/16) = 10240 blocks; layer0 writes Xbf (ELU), layer1 -> out
    attn_kernel<<<10240, 128, 0, stream>>>(HT, Fs, Fd, adj, elog, Xbf, out,
                                           (l == 1) ? 1 : 0);
  }
}
